// Block_22720376995910
// MI455X (gfx1250) — matrix operand order repaired and checked by reading, not compiled, not run on hardware
//
#include <hip/hip_runtime.h>
#include <hip/hip_bf16.h>
#include <math.h>

// ---------------------------------------------------------------------------
// CDNA5 (gfx1250) transformer block: LN -> QKV -> causal flash attention ->
// proj+residual -> LN -> router/top2 -> dense-all-experts MoE -> final sum.
// Matrix math: v_wmma_f32_16x16x32_bf16 (wave32). A-tiles staged to LDS with
// the Tensor Data Mover (tensor_load_to_lds + s_wait_tensorcnt); B-tiles
// converted fp32->bf16 into fragment-major LDS for vector ds reads.
// ---------------------------------------------------------------------------

typedef __attribute__((ext_vector_type(16))) __bf16 v16bf;
typedef __attribute__((ext_vector_type(8)))  __bf16 v8bf;
typedef __attribute__((ext_vector_type(8)))  float  v8f;
typedef unsigned int __attribute__((ext_vector_type(4))) u32x4;
typedef int  __attribute__((ext_vector_type(8))) i32x8;
typedef int  __attribute__((ext_vector_type(4))) i32x4;

#define B_S   4
#define T_S   1024
#define C_S   768
#define H_S   12
#define HD_S  64
#define E_S   16
#define DFF_S 3072
#define M_S   (B_S * T_S)   // 4096 tokens

// --------------------------- WMMA helpers ----------------------------------

__device__ __forceinline__ v8f wmma_bf16(v16bf a, v16bf b, v8f c) {
  // D = A(16x32) * B(32x16) + C(16x16), fp32 accumulate
  return __builtin_amdgcn_wmma_f32_16x16x32_bf16(false, a, false, b,
                                                 (short)0, c, false, false);
}

// A-fragment (16x32 bf16 MxK) from a row-major LDS tile. Per-lane data is two
// contiguous 16B runs: K = kh..kh+7 and kh+8..kh+15 (kh = (lane>>4)*16).
__device__ __forceinline__ v16bf load_a_frag_rm(const __bf16* s, int row0,
                                                int ld, int k0, int lane) {
  const int m  = lane & 15;
  const int kh = (lane >> 4) << 4;
  const __bf16* p = s + (row0 + m) * ld + k0 + kh;
  const v8bf lo = *(const v8bf*)p;         // K kh..kh+7
  const v8bf hi = *(const v8bf*)(p + 8);   // K kh+8..kh+15
  v16bf f;
#pragma unroll
  for (int t = 0; t < 8; ++t) { f[t] = lo[t]; f[t + 8] = hi[t]; }
  return f;
}

// B-fragment from fragment-major LDS: one contiguous 32B read per lane.
// Layout: Bs[((kchunk*8 + ntile)*32 + lane)*16 + t],
// value = W[kchunk*32 + (lane>>4)*16 + t][ntile*16 + (lane&15)].
__device__ __forceinline__ v16bf load_b_fragm(const __bf16* s, int kchunk,
                                              int ntile, int lane) {
  return *(const v16bf*)(s + ((((kchunk << 3) + ntile) << 5) + lane) * 16);
}

// ------------------------ Tensor Data Mover (TDM) ---------------------------
// Issue a 2D bf16 tensor_load_to_lds: tile 64 (dim0) x 16 (dim1) elements,
// 16B LDS padding per 128B row (pad_interval=32 dwords, pad_amount=4 dwords)
// => LDS row stride 72 elements. D# fields per CDNA5 ISA.

__device__ __forceinline__ void tdm_load_a_tile(const __bf16* gaddr,
                                                unsigned lds_off, int K_elems,
                                                int M_rows) {
  const unsigned long long ga = (unsigned long long)(uintptr_t)gaddr;
  u32x4 g0;
  g0[0] = 1u;                                    // count=1 (valid), user mode
  g0[1] = lds_off;                               // lds_addr (bytes)
  g0[2] = (unsigned)(ga & 0xffffffffu);          // global_addr[31:0]
  g0[3] = (unsigned)((ga >> 32) & 0x1ffffffu) |  // global_addr[56:32]
          (2u << 30);                            // type=2 ("image")
  i32x8 g1;
  g1[0] = (int)((1u << 16) |                     // data_size=1 -> 2 bytes
                (1u << 20) |                     // pad_enable
                (4u << 22) |                     // pad_interval: 32 dwords
                (3u << 25));                     // pad_amount: 4 dwords (16B)
  g1[1] = (int)(((unsigned)K_elems & 0xffffu) << 16);       // tensor_dim0 lo16
  g1[2] = (int)(((unsigned)K_elems >> 16) |                 // tensor_dim0 hi16
                (((unsigned)M_rows & 0xffffu) << 16));      // tensor_dim1 lo16
  g1[3] = (int)(((unsigned)M_rows >> 16) | (64u << 16));    // tile_dim0 = 64
  g1[4] = 16;                                    // tile_dim1=16, tile_dim2=0
  g1[5] = K_elems;                               // tensor_dim0_stride lo32
  g1[6] = 0;
  g1[7] = 0;
  const i32x4 gz = {0, 0, 0, 0};
#if defined(__clang_major__) && (__clang_major__ >= 23)
  const i32x8 gz8 = {0, 0, 0, 0, 0, 0, 0, 0};
  __builtin_amdgcn_tensor_load_to_lds(g0, g1, gz, gz, gz8, 0);
#else
  __builtin_amdgcn_tensor_load_to_lds(g0, g1, gz, gz, 0);
#endif
}

// --------------------------- GEMM kernel -----------------------------------
// Out[M,N] = epilogue(A_bf16[M,K] @ W_f32[K,N] + bias[N])
// 256 threads = 8 wave32 waves; block tile 128x128, K-step 64.
// A tile: each wave TDM-loads its own 16x64 slice. W tile: fp32->bf16 convert
// into fragment-major LDS. Each wave owns a 32x64 output sub-tile (2x4 frags).

enum { EPI_QKV = 0, EPI_RESID = 1, EPI_GELU = 2, EPI_MOE = 3 };

template <int EPI>
__global__ __launch_bounds__(256) void gemm_bf16_kernel(
    const __bf16* __restrict__ A, const float* __restrict__ W,
    const float* __restrict__ bias, float* __restrict__ outf,
    __bf16* __restrict__ outb, const float* __restrict__ resid,
    const float* __restrict__ gate, int gate_ld, int M, int N, int K,
    int accumulate) {
  constexpr int BM = 128, BN = 128, BK = 64;
  constexpr int LDA = BK + 8;  // 72 elements: TDM pad spreads banks
  __shared__ __bf16 As[BM * LDA];      // row-major, TDM-filled
  __shared__ __bf16 Bs[BK * BN];       // fragment-major, VALU-filled

  const int tid  = threadIdx.x;
  const int lane = tid & 31;
  const int wid  = tid >> 5;
  const int wm   = (wid >> 1) << 5;  // 0,32,64,96
  const int wn   = (wid & 1) << 6;   // 0,64
  const int m0   = blockIdx.y * BM;
  const int n0   = blockIdx.x * BN;
  const unsigned as_base = (unsigned)(uintptr_t)&As[0];

  v8f acc[2][4];
#pragma unroll
  for (int i = 0; i < 2; ++i)
#pragma unroll
    for (int j = 0; j < 4; ++j)
#pragma unroll
      for (int r = 0; r < 8; ++r) acc[i][j][r] = 0.0f;

  for (int k0 = 0; k0 < K; k0 += BK) {
    // --- A tile via Tensor Data Mover: wave w loads rows [w*16, w*16+16) ---
    tdm_load_a_tile(A + (size_t)(m0 + (wid << 4)) * K + k0,
                    as_base + (unsigned)(wid * 16 * LDA * 2), K, M);

    // --- W tile: fp32 -> bf16, scattered into fragment-major layout ---
#pragma unroll
    for (int it = 0; it < 8; ++it) {
      const int v  = tid + it * 256;
      const int r  = v >> 5;          // K row in tile: 0..63
      const int cs = (v & 31) << 2;   // N col: 0..124 step 4
      const float4 d = *(const float4*)(W + (size_t)(k0 + r) * N + n0 + cs);
      const int kchunk = r >> 5;
      const int khalf  = (r >> 4) & 1;
      const int t      = r & 15;
      const int ntile  = cs >> 4;
      const int nb     = cs & 15;
      __bf16* q = &Bs[(((((kchunk << 3) + ntile) << 5) + (khalf << 4) + nb) << 4) + t];
      q[0 * 16] = (__bf16)d.x;
      q[1 * 16] = (__bf16)d.y;
      q[2 * 16] = (__bf16)d.z;
      q[3 * 16] = (__bf16)d.w;
    }
    if (k0 + BK < K)  // global_prefetch_b8 next W tile
      __builtin_prefetch(W + (size_t)(k0 + BK + (tid >> 2)) * N + n0, 0, 0);

    __builtin_amdgcn_s_wait_tensorcnt(0);  // our wave's TDM slice landed
    __syncthreads();

#pragma unroll
    for (int ks = 0; ks < BK; ks += 32) {
      v16bf af[2];
#pragma unroll
      for (int i = 0; i < 2; ++i)
        af[i] = load_a_frag_rm(As, wm + (i << 4), LDA, ks, lane);
      v16bf bfr[4];
#pragma unroll
      for (int j = 0; j < 4; ++j)
        bfr[j] = load_b_fragm(Bs, ks >> 5, (wn >> 4) + j, lane);
#pragma unroll
      for (int i = 0; i < 2; ++i)
#pragma unroll
        for (int j = 0; j < 4; ++j)
          acc[i][j] = wmma_bf16(af[i], bfr[j], acc[i][j]);
    }
    __syncthreads();
  }

  // epilogue. D layout: VGPR r -> row r (lanes 0-15) / r+8 (lanes 16-31).
  const int mh = (lane < 16) ? 0 : 8;
  const int nl = lane & 15;
#pragma unroll
  for (int i = 0; i < 2; ++i) {
#pragma unroll
    for (int j = 0; j < 4; ++j) {
      const int col  = n0 + wn + (j << 4) + nl;
      const float bv = bias[col];
#pragma unroll
      for (int r = 0; r < 8; ++r) {
        const int row    = m0 + wm + (i << 4) + mh + r;
        float v          = acc[i][j][r] + bv;
        const size_t idx = (size_t)row * N + col;
        if constexpr (EPI == EPI_QKV) {
          outb[idx] = (__bf16)v;
        } else if constexpr (EPI == EPI_RESID) {
          outf[idx] = v + resid[idx];
        } else if constexpr (EPI == EPI_GELU) {
          const float g = 0.5f * v * (1.0f + erff(v * 0.70710678118f));
          outb[idx] = (__bf16)g;
        } else {  // EPI_MOE: per-row gate scale, accumulate across experts
          const float gt  = gate[(size_t)row * gate_ld];
          const float val = gt * v;
          if (accumulate) outf[idx] += val; else outf[idx] = val;
        }
      }
    }
  }
}

// ------------------------ flash attention kernel ---------------------------
// One wave32 per (b,h, 16 query rows). Online softmax; S and P*V via WMMA.
// qkv is bf16 [B*T, 3C]; q at col h*64, k at C+h*64, v at 2C+h*64.

__global__ __launch_bounds__(32) void attn_kernel(const __bf16* __restrict__ qkv,
                                                  __bf16* __restrict__ y) {
  __shared__ __bf16 Pt[16 * 40];  // P tile bounce, stride 40 (16B-aligned rows)
  const int lane = threadIdx.x;
  const int qb = blockIdx.x;
  const int bh = blockIdx.y;
  const int b  = bh / H_S;
  const int h  = bh - b * H_S;
  const int t0 = qb << 4;
  const int C3 = 3 * C_S;
  const __bf16* qbase = qkv + (size_t)b * T_S * C3 + h * HD_S;
  const __bf16* kbase = qbase + C_S;
  const __bf16* vbase = qbase + 2 * C_S;

  // Q fragments (16 rows x 64 head dims = two 16x32 A-frags), from global.
  v16bf qa[2];
  {
    const int m  = lane & 15;
    const int kh = (lane >> 4) << 4;
    const __bf16* p = qbase + (size_t)(t0 + m) * C3;
#pragma unroll
    for (int c = 0; c < 2; ++c) {
      const v8bf lo = *(const v8bf*)(p + (c << 5) + kh);
      const v8bf hi = *(const v8bf*)(p + (c << 5) + kh + 8);
#pragma unroll
      for (int t = 0; t < 8; ++t) { qa[c][t] = lo[t]; qa[c][t + 8] = hi[t]; }
    }
  }

  v8f o[4];
  float mrow[8], lrow[8];
#pragma unroll
  for (int j = 0; j < 4; ++j)
#pragma unroll
    for (int r = 0; r < 8; ++r) o[j][r] = 0.0f;
#pragma unroll
  for (int r = 0; r < 8; ++r) { mrow[r] = -1e30f; lrow[r] = 0.0f; }

  const int mh  = (lane < 16) ? 0 : 8;
  const int nl  = lane & 15;
  const int kb2 = (lane >> 4) << 4;

  for (int kb = 0; kb < t0 + 16; kb += 32) {  // causal: keys <= t0+15
    v8f s[2];
#pragma unroll
    for (int j = 0; j < 2; ++j)
#pragma unroll
      for (int r = 0; r < 8; ++r) s[j][r] = 0.0f;

    // S = Q @ K^T : B[kk][n] = K[key kb+j*16+n][hd c*32+kk] (contiguous 16)
#pragma unroll
    for (int c = 0; c < 2; ++c)
#pragma unroll
      for (int j = 0; j < 2; ++j) {
        const __bf16* p =
            kbase + (size_t)(kb + (j << 4) + nl) * C3 + (c << 5) + kb2;
        const v16bf kf = *(const v16bf*)p;
        s[j] = wmma_bf16(qa[c], kf, s[j]);
      }

    // scale + causal mask
#pragma unroll
    for (int j = 0; j < 2; ++j)
#pragma unroll
      for (int r = 0; r < 8; ++r) {
        const int col = kb + (j << 4) + nl;
        const int row = t0 + mh + r;
        float sv = s[j][r] * 0.125f;  // 1/sqrt(64)
        if (col > row) sv = -1e30f;
        s[j][r] = sv;
      }

    // online softmax: per-row reductions across 16-lane halves (D layout)
#pragma unroll
    for (int r = 0; r < 8; ++r) {
      float rm = fmaxf(s[0][r], s[1][r]);
#pragma unroll
      for (int d = 8; d >= 1; d >>= 1) rm = fmaxf(rm, __shfl_xor(rm, d, 32));
      const float mn    = fmaxf(mrow[r], rm);
      const float alpha = __expf(mrow[r] - mn);
      const float p0 = __expf(s[0][r] - mn);
      const float p1 = __expf(s[1][r] - mn);
      s[0][r] = p0; s[1][r] = p1;
      float rs = p0 + p1;
#pragma unroll
      for (int d = 8; d >= 1; d >>= 1) rs += __shfl_xor(rs, d, 32);
      lrow[r] = lrow[r] * alpha + rs;
      mrow[r] = mn;
#pragma unroll
      for (int j = 0; j < 4; ++j) o[j][r] *= alpha;
    }

    // bounce P through LDS to convert D-layout -> A-layout
#pragma unroll
    for (int j = 0; j < 2; ++j)
#pragma unroll
      for (int r = 0; r < 8; ++r)
        Pt[(mh + r) * 40 + (j << 4) + nl] = (__bf16)s[j][r];
    __syncthreads();
    const v16bf pa = load_a_frag_rm(Pt, 0, 40, 0, lane);

    // O += P @ V : B[kk][n] = V[key kb+kk][hd j*16+n]
#pragma unroll
    for (int j = 0; j < 4; ++j) {
      v16bf vf;
#pragma unroll
      for (int t = 0; t < 16; ++t)
        vf[t] = vbase[(size_t)(kb + kb2 + t) * C3 + (j << 4) + nl];
      o[j] = wmma_bf16(pa, vf, o[j]);
    }
    __syncthreads();
  }

#pragma unroll
  for (int j = 0; j < 4; ++j)
#pragma unroll
    for (int r = 0; r < 8; ++r) {
      const int row   = t0 + mh + r;
      const float val = o[j][r] / lrow[r];
      y[(size_t)(b * T_S + row) * C_S + h * HD_S + (j << 4) + nl] = (__bf16)val;
    }
}

// ----------------------------- LayerNorm -----------------------------------

__global__ __launch_bounds__(256) void ln_kernel(
    const float* __restrict__ x, const float* __restrict__ w,
    const float* __restrict__ b, float* __restrict__ outf,
    __bf16* __restrict__ outb) {
  const int row  = blockIdx.x;
  const int tid  = threadIdx.x;
  const int lane = tid & 31;
  const int wid  = tid >> 5;
  const float* xr = x + (size_t)row * C_S;
  float s = 0.0f, s2 = 0.0f;
  for (int c = tid; c < C_S; c += 256) {
    const float v = xr[c];
    s += v; s2 += v * v;
  }
#pragma unroll
  for (int d = 16; d >= 1; d >>= 1) {
    s  += __shfl_xor(s, d, 32);
    s2 += __shfl_xor(s2, d, 32);
  }
  __shared__ float red[16];
  __shared__ float stats[2];
  if (lane == 0) { red[wid] = s; red[wid + 8] = s2; }
  __syncthreads();
  if (tid == 0) {
    float ts = 0.0f, ts2 = 0.0f;
#pragma unroll
    for (int i = 0; i < 8; ++i) { ts += red[i]; ts2 += red[i + 8]; }
    const float mu  = ts / (float)C_S;
    const float var = ts2 / (float)C_S - mu * mu;
    stats[0] = mu;
    stats[1] = rsqrtf(var + 1e-5f);
  }
  __syncthreads();
  const float mu = stats[0], rv = stats[1];
  for (int c = tid; c < C_S; c += 256) {
    const float v = (xr[c] - mu) * rv * w[c] + b[c];
    if (outf) outf[(size_t)row * C_S + c] = v;
    outb[(size_t)row * C_S + c] = (__bf16)v;
  }
}

// -------------------------- router + top-2 gating --------------------------

__global__ __launch_bounds__(256) void router_kernel(
    const float* __restrict__ xn2, const float* __restrict__ Wr,
    float* __restrict__ gating) {
  const int row = blockIdx.x;
  const int tid = threadIdx.x;
  const int e   = tid >> 4;
  const int c0  = tid & 15;
  const float* xr = xn2 + (size_t)row * C_S;
  float p = 0.0f;
  for (int c = c0; c < C_S; c += 16) p += xr[c] * Wr[(size_t)c * E_S + e];
#pragma unroll
  for (int d = 8; d >= 1; d >>= 1) p += __shfl_xor(p, d, 32);
  __shared__ float logits[16];
  if (c0 == 0) logits[e] = p;
  __syncthreads();
  if (tid == 0) {
    float mx = -1e30f;
    for (int i = 0; i < 16; ++i) mx = fmaxf(mx, logits[i]);
    float pe[16]; float den = 0.0f;
    for (int i = 0; i < 16; ++i) { pe[i] = __expf(logits[i] - mx); den += pe[i]; }
    int i1 = 0;
    for (int i = 1; i < 16; ++i) if (pe[i] > pe[i1]) i1 = i;
    int i2 = (i1 == 0) ? 1 : 0;
    for (int i = 0; i < 16; ++i) if (i != i1 && pe[i] > pe[i2]) i2 = i;
    for (int i = 0; i < 16; ++i)
      gating[(size_t)row * E_S + i] = (i == i1 || i == i2) ? pe[i] / den : 0.0f;
  }
}

// ------------------------------ final sum ----------------------------------

__global__ __launch_bounds__(256) void add3_kernel(const float* __restrict__ a,
                                                   const float* __restrict__ b,
                                                   const float* __restrict__ c,
                                                   float* __restrict__ o, int n) {
  const int i = blockIdx.x * 256 + threadIdx.x;
  if (i < n) o[i] = a[i] + b[i] + c[i];
}

// ------------------------------- driver ------------------------------------

extern "C" void kernel_launch(void* const* d_in, const int* in_sizes, int n_in,
                              void* d_out, int out_size, void* d_ws,
                              size_t ws_size, hipStream_t stream) {
  (void)in_sizes; (void)n_in; (void)out_size; (void)ws_size;
  const float* x        = (const float*)d_in[0];
  const float* ln1_w    = (const float*)d_in[1];
  const float* ln1_b    = (const float*)d_in[2];
  const float* W_attn   = (const float*)d_in[3];
  const float* b_attn   = (const float*)d_in[4];
  const float* W_proj   = (const float*)d_in[5];
  const float* b_proj   = (const float*)d_in[6];
  const float* ln2_w    = (const float*)d_in[7];
  const float* ln2_b    = (const float*)d_in[8];
  const float* W_router = (const float*)d_in[9];
  const float* W1       = (const float*)d_in[10];
  const float* b1       = (const float*)d_in[11];
  const float* W2       = (const float*)d_in[12];
  const float* b2       = (const float*)d_in[13];
  float* out = (float*)d_out;

  char* ws = (char*)d_ws;
  size_t off = 0;
  auto alloc = [&](size_t bytes) -> void* {
    void* p = ws + off;
    off = (off + bytes + 255) & ~(size_t)255;
    return p;
  };
  __bf16* xn_b   = (__bf16*)alloc((size_t)M_S * C_S * 2);
  __bf16* qkv_b  = (__bf16*)alloc((size_t)M_S * 3 * C_S * 2);
  __bf16* y_b    = (__bf16*)alloc((size_t)M_S * C_S * 2);
  float*  x1     = (float*)alloc((size_t)M_S * C_S * 4);
  float*  xn2    = (float*)alloc((size_t)M_S * C_S * 4);
  __bf16* xn2_b  = (__bf16*)alloc((size_t)M_S * C_S * 2);
  float*  gating = (float*)alloc((size_t)M_S * E_S * 4);
  __bf16* h_b    = (__bf16*)alloc((size_t)M_S * DFF_S * 2);
  float*  ymoe   = (float*)alloc((size_t)M_S * C_S * 4);

  // 1) LN1 -> bf16
  ln_kernel<<<M_S, 256, 0, stream>>>(x, ln1_w, ln1_b, nullptr, xn_b);
  // 2) QKV GEMM (4096x2304x768)
  gemm_bf16_kernel<EPI_QKV><<<dim3(3 * C_S / 128, M_S / 128), 256, 0, stream>>>(
      xn_b, W_attn, b_attn, nullptr, qkv_b, nullptr, nullptr, 0, M_S, 3 * C_S,
      C_S, 0);
  // 3) causal flash attention
  attn_kernel<<<dim3(T_S / 16, B_S * H_S), 32, 0, stream>>>(qkv_b, y_b);
  // 4) proj + residual -> x1
  gemm_bf16_kernel<EPI_RESID><<<dim3(C_S / 128, M_S / 128), 256, 0, stream>>>(
      y_b, W_proj, b_proj, x1, nullptr, x, nullptr, 0, M_S, C_S, C_S, 0);
  // 5) LN2 -> fp32 + bf16
  ln_kernel<<<M_S, 256, 0, stream>>>(x1, ln2_w, ln2_b, xn2, xn2_b);
  // 6) router softmax + top-2 gating
  router_kernel<<<M_S, 256, 0, stream>>>(xn2, W_router, gating);
  // 7) dense all-experts MoE
  for (int e = 0; e < E_S; ++e) {
    gemm_bf16_kernel<EPI_GELU><<<dim3(DFF_S / 128, M_S / 128), 256, 0, stream>>>(
        xn2_b, W1 + (size_t)e * C_S * DFF_S, b1 + (size_t)e * DFF_S, nullptr,
        h_b, nullptr, nullptr, 0, M_S, DFF_S, C_S, 0);
    gemm_bf16_kernel<EPI_MOE><<<dim3(C_S / 128, M_S / 128), 256, 0, stream>>>(
        h_b, W2 + (size_t)e * DFF_S * C_S, b2 + (size_t)e * C_S, ymoe, nullptr,
        nullptr, gating + e, E_S, M_S, C_S, DFF_S, (e == 0) ? 0 : 1);
  }
  // 8) out = x1 + xn2 + y_moe
  const int ntot = M_S * C_S;
  add3_kernel<<<(ntot + 255) / 256, 256, 0, stream>>>(x1, xn2, ymoe, out, ntot);
}
